// Net_4432406249735
// MI455X (gfx1250) — compile-verified, matrix operand order checked by reading, not run on hardware
//
#include <hip/hip_runtime.h>
#include <hip/hip_bf16.h>

typedef __attribute__((ext_vector_type(16))) __bf16 v16bf;
typedef __attribute__((ext_vector_type(8)))  __bf16 v8bf;
typedef __attribute__((ext_vector_type(8)))  float  v8f;

#define FINAL 160000      // final output length
#define L0LEN 165110      // length after causal conv
#define KALL  1440        // 45 layers * 32 channels (deferred skip GEMM K)

// ---------------------------------------------------------------------------
// WMMA helper: D = A(16x32 bf16) * B(32x16 bf16) + C(16x16 f32)
// ---------------------------------------------------------------------------
__device__ __forceinline__ v8f wmma_bf16(v16bf a, v16bf b, v8f c) {
  return __builtin_amdgcn_wmma_f32_16x16x32_bf16(
      /*neg_a=*/false, a, /*neg_b=*/false, b,
      /*c_mod=*/(short)0, c, /*reuse_a=*/false, /*reuse_b=*/false);
}

// K-index mapping for 16-bit A/B fragments (ISA 7.12.2):
// VGPR0..3 hold K=0..7 (lanes 0-15) / K=8..15 (lanes 16-31),
// VGPR4..7 hold K=16..23 / K=24..31.
__device__ __forceinline__ int kidx(int j, int lane) {
  return (j < 8 ? j : j + 8) + ((lane >= 16) ? 8 : 0);
}

__device__ __forceinline__ float sigmf(float x) {
  return 1.0f / (1.0f + __expf(-x));
}

__device__ __forceinline__ v16bf cat8(v8bf lo, v8bf hh) {
  v16bf r;
#pragma unroll
  for (int j = 0; j < 8; ++j) { r[j] = lo[j]; r[j + 8] = hh[j]; }
  return r;
}

// Pre-packed A fragment: frag f is 32 lanes x 16 bf16, per-lane contiguous 32B.
__device__ __forceinline__ v16bf load_frag(const __bf16* __restrict__ P, int f, int lane) {
  return *(const v16bf*)(P + ((long)f * 32 + lane) * 16);
}

// B fragment from position-major bf16 activations act[pos][ldc]; the K-mapping
// {0..7,16..23}+8*hi turns a lane's 16 elements into two contiguous 16B chunks.
__device__ __forceinline__ v16bf load_b_pm(const __bf16* __restrict__ act, long ldc,
                                           long pos, int koff, int lane) {
  const __bf16* r = act + pos * ldc + koff + ((lane >= 16) ? 8 : 0);
  return cat8(*(const v8bf*)r, *(const v8bf*)(r + 16));
}

// ---------------------------------------------------------------------------
// Weight packing kernels (run once; ~3MB total, lives in L2)
// ---------------------------------------------------------------------------
// conv weights w[45][32][32][3] -> frag f=((i*3+k)*2+h)
__global__ void pack_conv_k(const float* __restrict__ W, __bf16* __restrict__ P) {
  int f = blockIdx.x, lane = threadIdx.x;
  int h = f & 1, k = (f >> 1) % 3, i = f / 6;
  int oc = h * 16 + (lane & 15);
#pragma unroll
  for (int j = 0; j < 16; ++j) {
    int ic = kidx(j, lane);
    P[((long)f * 32 + lane) * 16 + j] = (__bf16)W[((long)(i * 32 + oc) * 32 + ic) * 3 + k];
  }
}

// row-major W[M][ld] -> frag f = mh*nks + ks; rows mh*16.., cols ks*32..
__global__ void pack_rm_k(const float* __restrict__ W, __bf16* __restrict__ P,
                          long ld, int nks) {
  int f = blockIdx.x, lane = threadIdx.x;
  int mh = f / nks, ks = f % nks;
  long row = mh * 16 + (lane & 15);
#pragma unroll
  for (int j = 0; j < 16; ++j)
    P[((long)f * 32 + lane) * 16 + j] = (__bf16)W[row * ld + ks * 32 + kidx(j, lane)];
}

// ---------------------------------------------------------------------------
// Causal conv -> position-major fp32 + bf16 copies. Consecutive threads cover
// consecutive channels of one position => fully coalesced stores.
// ---------------------------------------------------------------------------
__global__ void causal_conv_k(const float* __restrict__ xin,
                              const float* __restrict__ wc,
                              const float* __restrict__ bc,
                              float* __restrict__ xf0, __bf16* __restrict__ xb0) {
  long idx = (long)blockIdx.x * blockDim.x + threadIdx.x;
  if (idx >= 32L * L0LEN) return;
  int  c = (int)(idx & 31);
  long p = idx >> 5;
  float acc = bc[c];
#pragma unroll
  for (int t = 0; t < 33; ++t) acc += wc[c * 33 + t] * xin[p + t];
  xf0[p * 32 + c] = acc;
  xb0[p * 32 + c] = (__bf16)acc;
}

__global__ void bias_sum_k(const float* __restrict__ bskip, float* __restrict__ bsum) {
  int oc = blockIdx.x * blockDim.x + threadIdx.x;
  if (oc >= 512) return;
  float s = 0.f;
  for (int i = 0; i < 45; ++i) s += bskip[i * 512 + oc];
  bsum[oc] = s;
}

// ---------------------------------------------------------------------------
// One dilated gated residual block. Wave = 32 out-channels x 16 positions.
//   gate GEMMs: M=32, K=96 (3 taps x 32ch) -> 12 wmma (B shared tanh/sigmoid)
//   dense GEMM: M=32, K=32 (+bias+residual) -> 2 wmma, B re-laid out via LDS
// ---------------------------------------------------------------------------
__global__ void __launch_bounds__(128) wavenet_block_k(
    const float* __restrict__ xf, const __bf16* __restrict__ xb,
    float* __restrict__ yf, __bf16* __restrict__ yb,
    const __bf16* __restrict__ PTi, const __bf16* __restrict__ PSi,
    const __bf16* __restrict__ PDi,
    const float* __restrict__ bt, const float* __restrict__ bsg,
    const float* __restrict__ bd,
    __bf16* __restrict__ xall_col,   // Xall + i*32 (position-major, ld=KALL)
    int Lz, int d, int cut) {
  __shared__ __bf16 zt[4][16][32];

  int lane = threadIdx.x & 31;
  int w    = threadIdx.x >> 5;
  int p0   = (blockIdx.x * 4 + w) * 16;
  int n    = lane & 15;
  int hi   = (lane >= 16) ? 8 : 0;
  int p    = p0 + n;
  int pc   = (p < Lz) ? p : (Lz - 1);   // clamp; stores guarded below
  bool inL = p < Lz;

  v8f ct0 = *(const v8f*)(bt + hi),       ct1 = *(const v8f*)(bt + 16 + hi);
  v8f cs0 = *(const v8f*)(bsg + hi),      cs1 = *(const v8f*)(bsg + 16 + hi);

#pragma unroll
  for (int k = 0; k < 3; ++k) {
    v16bf b = load_b_pm(xb, 32, (long)pc + (long)k * d, 0, lane);
    ct0 = wmma_bf16(load_frag(PTi, k * 2 + 0, lane), b, ct0);
    ct1 = wmma_bf16(load_frag(PTi, k * 2 + 1, lane), b, ct1);
    cs0 = wmma_bf16(load_frag(PSi, k * 2 + 0, lane), b, cs0);
    cs1 = wmma_bf16(load_frag(PSi, k * 2 + 1, lane), b, cs1);
  }

  // gate in fp32; z tile -> LDS (position-major) + export window into X_all
  v8bf z0v, z1v;
#pragma unroll
  for (int r = 0; r < 8; ++r) {
    z0v[r] = (__bf16)(tanhf(ct0[r]) * sigmf(cs0[r]));
    z1v[r] = (__bf16)(tanhf(ct1[r]) * sigmf(cs1[r]));
  }
  *(v8bf*)(&zt[w][n][hi])      = z0v;
  *(v8bf*)(&zt[w][n][16 + hi]) = z1v;
  int ps = p - cut;
  if (inL && ps >= 0 && ps < FINAL) {
    *(v8bf*)(xall_col + (long)ps * KALL + hi)      = z0v;
    *(v8bf*)(xall_col + (long)ps * KALL + 16 + hi) = z1v;
  }
  __syncthreads();

  // dense 1x1: C init = bias + residual xin = x[:, p+d]
  const float* rp = xf + (long)(pc + d) * 32;
  v8f cd0 = *(const v8f*)(bd + hi)      + *(const v8f*)(rp + hi);
  v8f cd1 = *(const v8f*)(bd + 16 + hi) + *(const v8f*)(rp + 16 + hi);

  const __bf16* zr = &zt[w][n][hi];
  v16bf bz = cat8(*(const v8bf*)zr, *(const v8bf*)(zr + 16));
  cd0 = wmma_bf16(load_frag(PDi, 0, lane), bz, cd0);
  cd1 = wmma_bf16(load_frag(PDi, 1, lane), bz, cd1);

  if (inL) {
    float* op = yf + (long)p * 32;
    *(v8f*)(op + hi)      = cd0;
    *(v8f*)(op + 16 + hi) = cd1;
    v8bf cb0, cb1;
#pragma unroll
    for (int r = 0; r < 8; ++r) { cb0[r] = (__bf16)cd0[r]; cb1[r] = (__bf16)cd1[r]; }
    __bf16* ob = yb + (long)p * 32;
    *(v8bf*)(ob + hi)      = cb0;
    *(v8bf*)(ob + 16 + hi) = cb1;
  }
}

// ---------------------------------------------------------------------------
// Deferred skip GEMM: H0 = relu( W_skip[512 x 1440] @ X_all^T + bsum )
// X_all is position-major [FINAL][1440]; one pass over 460MB instead of 45
// read-modify-writes of a 327MB fp32 accumulator.
// ---------------------------------------------------------------------------
__global__ void __launch_bounds__(128) skip_gemm_k(
    const __bf16* __restrict__ Xall, const __bf16* __restrict__ PSK,
    const float* __restrict__ bsum, __bf16* __restrict__ H0) {
  int lane = threadIdx.x & 31;
  int w    = threadIdx.x >> 5;
  long n   = (long)(blockIdx.x * 4 + w) * 16 + (lane & 15);
  int ocb  = blockIdx.y * 32;
  int mh   = blockIdx.y * 2;
  int hi   = (lane >= 16) ? 8 : 0;

  v8f c0 = *(const v8f*)(bsum + ocb + hi);
  v8f c1 = *(const v8f*)(bsum + ocb + 16 + hi);
  for (int i = 0; i < 45; ++i) {
    v16bf b = load_b_pm(Xall, KALL, n, i * 32, lane);
    c0 = wmma_bf16(load_frag(PSK, i * 32 + mh, lane), b, c0);
    c1 = wmma_bf16(load_frag(PSK, i * 32 + mh + 1, lane), b, c1);
  }
  v8bf o0, o1;
#pragma unroll
  for (int r = 0; r < 8; ++r) {
    o0[r] = (__bf16)fmaxf(c0[r], 0.f);
    o1[r] = (__bf16)fmaxf(c1[r], 0.f);
  }
  *(v8bf*)(H0 + n * 512 + ocb + hi)      = o0;
  *(v8bf*)(H0 + n * 512 + ocb + 16 + hi) = o1;
}

// ---------------------------------------------------------------------------
// Post 1x1 GEMM: out = act( W[M x 512] @ Bin^T + bias ), Bin position-major.
// RELU_BF16: relu -> bf16 position-major (intermediate); else fp32 channel-
// major final output (matches reference NCH layout).
// ---------------------------------------------------------------------------
template <bool RELU_BF16>
__global__ void __launch_bounds__(128) post_gemm_k(
    const __bf16* __restrict__ Bin, const __bf16* __restrict__ PW,
    const float* __restrict__ bias, void* __restrict__ outp) {
  int lane = threadIdx.x & 31;
  int w    = threadIdx.x >> 5;
  long n   = (long)(blockIdx.x * 4 + w) * 16 + (lane & 15);
  int ocb  = blockIdx.y * 32;
  int mh   = blockIdx.y * 2;
  int hi   = (lane >= 16) ? 8 : 0;

  v8f c0 = *(const v8f*)(bias + ocb + hi);
  v8f c1 = *(const v8f*)(bias + ocb + 16 + hi);
#pragma unroll 4
  for (int ks = 0; ks < 16; ++ks) {
    v16bf b = load_b_pm(Bin, 512, n, ks * 32, lane);
    c0 = wmma_bf16(load_frag(PW, mh * 16 + ks, lane), b, c0);
    c1 = wmma_bf16(load_frag(PW, (mh + 1) * 16 + ks, lane), b, c1);
  }
  if (RELU_BF16) {
    __bf16* o = (__bf16*)outp;
    v8bf o0, o1;
#pragma unroll
    for (int r = 0; r < 8; ++r) {
      o0[r] = (__bf16)fmaxf(c0[r], 0.f);
      o1[r] = (__bf16)fmaxf(c1[r], 0.f);
    }
    *(v8bf*)(o + n * 512 + ocb + hi)      = o0;
    *(v8bf*)(o + n * 512 + ocb + 16 + hi) = o1;
  } else {
    float* o = (float*)outp;
#pragma unroll
    for (int r = 0; r < 8; ++r) {
      o[(long)(ocb + r + hi) * FINAL + n]      = c0[r];
      o[(long)(ocb + 16 + r + hi) * FINAL + n] = c1[r];
    }
  }
}

// ---------------------------------------------------------------------------
// Host orchestration
// ---------------------------------------------------------------------------
extern "C" void kernel_launch(void* const* d_in, const int* in_sizes, int n_in,
                              void* d_out, int out_size, void* d_ws, size_t ws_size,
                              hipStream_t stream) {
  (void)in_sizes; (void)n_in; (void)out_size; (void)ws_size;
  const float* x_in     = (const float*)d_in[0];
  const float* w_causal = (const float*)d_in[1];
  const float* b_causal = (const float*)d_in[2];
  const float* w_tanh   = (const float*)d_in[3];
  const float* b_tanh   = (const float*)d_in[4];
  const float* w_sig    = (const float*)d_in[5];
  const float* b_sig    = (const float*)d_in[6];
  const float* w_skip   = (const float*)d_in[7];
  const float* b_skip   = (const float*)d_in[8];
  const float* w_dense  = (const float*)d_in[9];
  const float* b_dense  = (const float*)d_in[10];
  const float* w_post1  = (const float*)d_in[11];
  const float* b_post1  = (const float*)d_in[12];
  const float* w_post2  = (const float*)d_in[13];
  const float* b_post2  = (const float*)d_in[14];

  // workspace layout (all position-major activations)
  char* ws = (char*)d_ws;
  size_t off = 0;
  auto take = [&](size_t bytes) { void* p = ws + off; off += (bytes + 255) & ~size_t(255); return p; };
  float*  xfA  = (float*)take(32L * L0LEN * sizeof(float));
  float*  xfB  = (float*)take(32L * L0LEN * sizeof(float));
  __bf16* xbA  = (__bf16*)take(32L * L0LEN * sizeof(__bf16));
  __bf16* xbB  = (__bf16*)take(32L * L0LEN * sizeof(__bf16));
  __bf16* Xall = (__bf16*)take((long)FINAL * KALL * sizeof(__bf16));
  __bf16* H0   = (__bf16*)take((long)FINAL * 512 * sizeof(__bf16));
  __bf16* H1   = (__bf16*)take((long)FINAL * 512 * sizeof(__bf16));
  float*  bsum = (float*)take(512 * sizeof(float));
  __bf16* PT   = (__bf16*)take(270L * 512 * sizeof(__bf16));   // 45*3*2 frags
  __bf16* PS   = (__bf16*)take(270L * 512 * sizeof(__bf16));
  __bf16* PD   = (__bf16*)take(90L  * 512 * sizeof(__bf16));   // 45*2
  __bf16* PSK  = (__bf16*)take(1440L * 512 * sizeof(__bf16));  // 45*32
  __bf16* PP1  = (__bf16*)take(512L * 512 * sizeof(__bf16));   // 32 mh * 16 ks
  __bf16* PP2  = (__bf16*)take(256L * 512 * sizeof(__bf16));   // 16 mh * 16 ks

  // 0) pack weights into WMMA fragment layout (bf16)
  pack_conv_k<<<270, 32, 0, stream>>>(w_tanh, PT);
  pack_conv_k<<<270, 32, 0, stream>>>(w_sig, PS);
  pack_rm_k<<<90, 32, 0, stream>>>(w_dense, PD, 32, 1);
  pack_rm_k<<<1440, 32, 0, stream>>>(w_skip, PSK, 32, 1);
  pack_rm_k<<<512, 32, 0, stream>>>(w_post1, PP1, 512, 16);
  pack_rm_k<<<256, 32, 0, stream>>>(w_post2, PP2, 512, 16);
  bias_sum_k<<<2, 256, 0, stream>>>(b_skip, bsum);

  // 1) causal conv
  {
    long total = 32L * L0LEN;
    causal_conv_k<<<dim3((unsigned)((total + 255) / 256)), 256, 0, stream>>>(
        x_in, w_causal, b_causal, xfA, xbA);
  }

  // 2) 45 dilated blocks (ping-pong)
  int dil[45];
  for (int r5 = 0; r5 < 5; ++r5)
    for (int e = 0; e < 9; ++e) dil[r5 * 9 + e] = 1 << e;

  float*  curf = xfA; float*  nxtf = xfB;
  __bf16* curb = xbA; __bf16* nxtb = xbB;
  int Lin = L0LEN;
  for (int i = 0; i < 45; ++i) {
    int d   = dil[i];
    int Lz  = Lin - 2 * d;
    int cut = (Lz - FINAL) / 2;
    int blocks = ((Lz + 15) / 16 + 3) / 4;
    wavenet_block_k<<<dim3((unsigned)blocks), 128, 0, stream>>>(
        curf, curb, nxtf, nxtb,
        PT + (long)i * 6 * 512, PS + (long)i * 6 * 512, PD + (long)i * 2 * 512,
        b_tanh + (long)i * 32, b_sig + (long)i * 32, b_dense + (long)i * 32,
        Xall + (long)i * 32, Lz, d, cut);
    float*  tf = curf; curf = nxtf; nxtf = tf;
    __bf16* tb = curb; curb = nxtb; nxtb = tb;
    Lin = Lz;
  }

  // 3) deferred skip GEMM (K = 1440) + relu
  skip_gemm_k<<<dim3(FINAL / 64, 16), 128, 0, stream>>>(Xall, PSK, bsum, H0);

  // 4) post layers
  post_gemm_k<true><<<dim3(FINAL / 64, 16), 128, 0, stream>>>(H0, PP1, b_post1, H1);
  post_gemm_k<false><<<dim3(FINAL / 64, 8), 128, 0, stream>>>(H1, PP2, b_post2, d_out);
}
